// CCLoss_70729521430644
// MI455X (gfx1250) — matrix operand order repaired and checked by reading, not compiled, not run on hardware
//
#include <hip/hip_runtime.h>

// CDNA5 (gfx1250) wave32 WMMA types
typedef __attribute__((ext_vector_type(16))) _Float16 v16h;
typedef __attribute__((ext_vector_type(8)))  float    v8f;
typedef __attribute__((ext_vector_type(2)))  __fp16   fp16x2;   // return type of cvt_pkrtz

#define WAVES    4
#define H_IMG    512
#define W_IMG    512
#define NTILE_X  32
#define NTILE_Y  32
#define NIMG     16
#define NTILES   (NTILE_X * NTILE_Y * NIMG)   // 16384 16x16 tiles
#define NBLOCKS  (NTILES / WAVES)             // 4096 blocks of 128 threads

#define SX_PITCH 36          // halves per row of sI/sJ: 72B rows, 8B-aligned, odd bank stride
#define SH_PITCH 132         // halves per row of sH: 16 cols x 8 halves + 4 pad -> 264B rows

union FragH { v16h v; _Float16 h[16]; };
union FragF { v8f  v; float    f[8];  };
union P2    { fp16x2 h; unsigned u;   };
union P4    { uint2 u2; _Float16 h[4]; };

// One wave per 16x16 output tile. Separable 9x9 box sum done as banded
// 16x16x32 f16 WMMAs (K=32 covers the 24-wide halo window), f32 accumulate.
__global__ __launch_bounds__(128) void ncc_tile_kernel(const float* __restrict__ I,
                                                       const float* __restrict__ J,
                                                       float* __restrict__ partials)
{
    __shared__ _Float16 sI[WAVES][32 * SX_PITCH];   // 32x32 halo tile of I (f16)
    __shared__ _Float16 sJ[WAVES][32 * SX_PITCH];   // 32x32 halo tile of J
    __shared__ _Float16 sH[WAVES][32 * SH_PITCH];   // H sums: [row][col][ch0..ch4,pad]
    __shared__ float    red[128];

    const int tid  = threadIdx.x;
    const int wave = tid >> 5;
    const int lane = tid & 31;

    const int tileId = blockIdx.x * WAVES + wave;   // < 16384
    const int tx  = tileId & (NTILE_X - 1);
    const int ty  = (tileId >> 5) & (NTILE_Y - 1);
    const int img = tileId >> 10;

    const float* bI = I + (size_t)img * (H_IMG * W_IMG);
    const float* bJ = J + (size_t)img * (H_IMG * W_IMG);

    const int r0 = ty * 16 - 8;   // LDS row lr -> global row r0 + lr
    const int c0 = tx * 16 - 8;   // LDS col lc -> global col c0 + lc

    // warm L2/WGP$ for this tile (global_prefetch_b8)
    {
        int pr = r0 < 0 ? 0 : r0;
        int pc = c0 < 0 ? 0 : c0;
        __builtin_prefetch(bI + (size_t)pr * W_IMG + pc, 0, 0);
        __builtin_prefetch(bJ + (size_t)pr * W_IMG + pc, 0, 0);
    }

    const bool interior = (tx >= 1) && (tx <= 30) && (ty >= 1) && (ty <= 30);  // wave-uniform

    if (interior) {
        // Fast path: float4 loads, packed f16 stores, no bounds checks.
        const int chunk = lane & 7;        // float4 column chunk (8 per 32-wide row)
        const int rsub  = lane >> 3;       // 4 rows covered per iteration
#pragma unroll
        for (int it = 0; it < 8; ++it) {
            const int r = 4 * it + rsub;
            const float4 vi = *(const float4*)(bI + (size_t)(r0 + r) * W_IMG + c0 + chunk * 4);
            const float4 vj = *(const float4*)(bJ + (size_t)(r0 + r) * W_IMG + c0 + chunk * 4);
            P2 a0, a1, b0, b1;
            a0.h = __builtin_amdgcn_cvt_pkrtz(vi.x, vi.y);
            a1.h = __builtin_amdgcn_cvt_pkrtz(vi.z, vi.w);
            b0.h = __builtin_amdgcn_cvt_pkrtz(vj.x, vj.y);
            b1.h = __builtin_amdgcn_cvt_pkrtz(vj.z, vj.w);
            uint2 wi; wi.x = a0.u; wi.y = a1.u;
            uint2 wj; wj.x = b0.u; wj.y = b1.u;
            *(uint2*)&sI[wave][r * SX_PITCH + chunk * 4] = wi;
            *(uint2*)&sJ[wave][r * SX_PITCH + chunk * 4] = wj;
        }
    } else {
        // Border path: scalar, zero-padded (matches reduce_window zero pad).
        for (int e = lane; e < 32 * 32; e += 32) {
            int lr = e >> 5, lc = e & 31;
            int gr = r0 + lr, gc = c0 + lc;
            float vi = 0.0f, vj = 0.0f;
            if ((unsigned)gr < (unsigned)H_IMG && (unsigned)gc < (unsigned)W_IMG) {
                size_t idx = (size_t)gr * W_IMG + gc;
                vi = bI[idx];
                vj = bJ[idx];
            }
            sI[wave][lr * SX_PITCH + lc] = (_Float16)vi;
            sJ[wave][lr * SX_PITCH + lc] = (_Float16)vj;
        }
    }
    __syncthreads();

    const int n  = lane & 15;   // N for B/C/D layouts; M for A layout
    const int hi = lane >> 4;

    // Constant banded-ones fragments.
    //  bh (B layout, 32x16): B[k][n] = 1 iff n+4 <= k <= n+12  (horizontal 9-tap)
    //  av (A layout, 16x32): A[m][k] = 1 iff m+4 <= k <= m+12  (vertical 9-tap)
    FragH bh, av;
#pragma unroll
    for (int i = 0; i < 16; ++i) {
        int kB = (i < 8) ? (hi * 8 + i) : (16 + hi * 8 + (i - 8));   // B: same K striping as A
        bh.h[i] = (kB >= n + 4 && kB <= n + 12) ? (_Float16)1.0f : (_Float16)0.0f;
        int v = i >> 1, hf = i & 1;   // A: documented 16-bit A striping
        int kA = ((v < 4) ? 0 : 16) + ((v & 3) << 1) + hf + (hi ? 8 : 0);
        av.h[i] = (kA >= n + 4 && kA <= n + 12) ? (_Float16)1.0f : (_Float16)0.0f;
    }

    v8f cz = {};

    // Horizontal pass: two 16-row blocks -> H rows 0..31 for all 5 channels
#pragma unroll
    for (int rb = 0; rb < 2; ++rb) {
        FragH aI, aJ, aII, aJJ, aIJ;
#pragma unroll
        for (int i = 0; i < 16; ++i) {
            int v = i >> 1, hf = i & 1;
            int K = ((v < 4) ? 0 : 16) + ((v & 3) << 1) + hf + (hi ? 8 : 0);
            _Float16 xi = sI[wave][(rb * 16 + n) * SX_PITCH + K];
            _Float16 xj = sJ[wave][(rb * 16 + n) * SX_PITCH + K];
            aI.h[i]  = xi;
            aJ.h[i]  = xj;
            aII.h[i] = xi * xi;       // elementwise in A layout: free extra channels
            aJJ.h[i] = xj * xj;
            aIJ.h[i] = xi * xj;
        }
        FragF d0, d1, d2, d3, d4;
        d0.v = __builtin_amdgcn_wmma_f32_16x16x32_f16(false, aI.v,  false, bh.v, (short)0, cz, false, false);
        d1.v = __builtin_amdgcn_wmma_f32_16x16x32_f16(false, aJ.v,  false, bh.v, (short)0, cz, false, false);
        d2.v = __builtin_amdgcn_wmma_f32_16x16x32_f16(false, aII.v, false, bh.v, (short)0, cz, false, false);
        d3.v = __builtin_amdgcn_wmma_f32_16x16x32_f16(false, aJJ.v, false, bh.v, (short)0, cz, false, false);
        d4.v = __builtin_amdgcn_wmma_f32_16x16x32_f16(false, aIJ.v, false, bh.v, (short)0, cz, false, false);
#pragma unroll
        for (int g = 0; g < 8; ++g) {   // C/D layout: VGPR g -> row g (+8 for hi lanes)
            int hr   = rb * 16 + g + 8 * hi;
            int base = hr * SH_PITCH + n * 8;   // 8B-aligned channel bundle
            P2 c01, c23;
            c01.h = __builtin_amdgcn_cvt_pkrtz(d0.f[g], d1.f[g]);
            c23.h = __builtin_amdgcn_cvt_pkrtz(d2.f[g], d3.f[g]);
            uint2 w; w.x = c01.u; w.y = c23.u;
            *(uint2*)&sH[wave][base] = w;               // ch0..ch3, one b64
            sH[wave][base + 4] = (_Float16)d4.f[g];     // ch4
        }
    }
    __syncthreads();

    // Vertical pass: gather H rows in B layout (bank-conflict-free column walk)
    FragH bv0, bv1, bv2, bv3, bv4;
#pragma unroll
    for (int i = 0; i < 16; ++i) {
        int k    = (i < 8) ? (hi * 8 + i) : (16 + hi * 8 + (i - 8));
        int base = k * SH_PITCH + n * 8;
        P4 w; w.u2 = *(const uint2*)&sH[wave][base];
        bv0.h[i] = w.h[0];
        bv1.h[i] = w.h[1];
        bv2.h[i] = w.h[2];
        bv3.h[i] = w.h[3];
        bv4.h[i] = sH[wave][base + 4];
    }
    FragF s0, s1, s2, s3, s4;
    s0.v = __builtin_amdgcn_wmma_f32_16x16x32_f16(false, av.v, false, bv0.v, (short)0, cz, false, false);
    s1.v = __builtin_amdgcn_wmma_f32_16x16x32_f16(false, av.v, false, bv1.v, (short)0, cz, false, false);
    s2.v = __builtin_amdgcn_wmma_f32_16x16x32_f16(false, av.v, false, bv2.v, (short)0, cz, false, false);
    s3.v = __builtin_amdgcn_wmma_f32_16x16x32_f16(false, av.v, false, bv3.v, (short)0, cz, false, false);
    s4.v = __builtin_amdgcn_wmma_f32_16x16x32_f16(false, av.v, false, bv4.v, (short)0, cz, false, false);

    // Per-pixel cc (algebraically identical to the reference formulas)
    const float inv81 = 1.0f / 81.0f;
    float acc = 0.0f;
#pragma unroll
    for (int g = 0; g < 8; ++g) {
        float Is = s0.f[g], Js = s1.f[g];
        float I2 = s2.f[g], J2 = s3.f[g], IJ = s4.f[g];
        float cross = fmaxf(IJ - Is * Js * inv81, 1e-5f);
        float Iv    = fmaxf(I2 - Is * Is * inv81, 1e-5f);
        float Jv    = fmaxf(J2 - Js * Js * inv81, 1e-5f);
        acc += (cross * cross) * __builtin_amdgcn_rcpf(Iv * Jv);
    }

    // Deterministic block reduction -> one partial per block
    red[tid] = acc;
    __syncthreads();
#pragma unroll
    for (int off = 64; off > 0; off >>= 1) {
        if (tid < off) red[tid] += red[tid + off];
        __syncthreads();
    }
    if (tid == 0) partials[blockIdx.x] = red[0];
}

__global__ __launch_bounds__(256) void ncc_finalize_kernel(const float* __restrict__ partials,
                                                           int n, float* __restrict__ out)
{
    __shared__ float red[256];
    float s = 0.0f;
    for (int i = threadIdx.x; i < n; i += 256) s += partials[i];
    red[threadIdx.x] = s;
    __syncthreads();
#pragma unroll
    for (int off = 128; off > 0; off >>= 1) {
        if (threadIdx.x < off) red[threadIdx.x] += red[threadIdx.x + off];
        __syncthreads();
    }
    if (threadIdx.x == 0)
        out[0] = -red[0] / (float)(NIMG * H_IMG * W_IMG);   // -mean(cc)
}

extern "C" void kernel_launch(void* const* d_in, const int* in_sizes, int n_in,
                              void* d_out, int out_size, void* d_ws, size_t ws_size,
                              hipStream_t stream)
{
    (void)in_sizes; (void)n_in; (void)out_size; (void)ws_size;
    const float* y_true = (const float*)d_in[0];
    const float* y_pred = (const float*)d_in[1];
    float* partials = (float*)d_ws;              // NBLOCKS floats = 16 KB scratch
    float* out      = (float*)d_out;

    ncc_tile_kernel<<<NBLOCKS, 128, 0, stream>>>(y_true, y_pred, partials);
    ncc_finalize_kernel<<<1, 256, 0, stream>>>(partials, NBLOCKS, out);
}
